// TriangleMultiplication_7232724927143
// MI455X (gfx1250) — matrix operand order repaired and checked by reading, compiled, not run on hardware
//
#include <hip/hip_runtime.h>
#include <hip/hip_bf16.h>
#include <math.h>

#define NRES 768
#define CZ   128
#define NN   (NRES*NRES)

#if defined(__has_builtin)
#  if __has_builtin(__builtin_amdgcn_global_load_async_to_lds_b128)
#    define HAVE_ASYNC_LDS 1
#  endif
#endif
#ifndef HAVE_ASYNC_LDS
#  define HAVE_ASYNC_LDS 0
#endif

typedef __attribute__((ext_vector_type(16))) __bf16 v16bf;
typedef __attribute__((ext_vector_type(8)))  float  v8f;

union FragBF { v16bf v; unsigned int u[8]; uint4 q[2]; };
static_assert(sizeof(FragBF) == 32, "frag size");

__device__ __forceinline__ unsigned short f2bf(float f) {
  union { float f; unsigned int u; } x; x.f = f;
  unsigned int r = x.u + 0x7FFFu + ((x.u >> 16) & 1u);
  return (unsigned short)(r >> 16);
}
__device__ __forceinline__ float bf2f(unsigned short h) {
  union { unsigned int u; float f; } x; x.u = ((unsigned int)h) << 16; return x.f;
}
__device__ __forceinline__ float sigmoidf_(float x) { return 1.0f / (1.0f + __expf(-x)); }
__device__ __forceinline__ unsigned int pack2(float a, float b) {
  return (unsigned int)f2bf(a) | ((unsigned int)f2bf(b) << 16);
}

#if HAVE_ASYNC_LDS
// Matches the builtin's parameter pointee type exactly (per compiler diagnostic):
// int vector_size(16), global (AS1) source, LDS (AS3) destination.
typedef int v4i_ls __attribute__((vector_size(16)));
typedef v4i_ls __attribute__((address_space(1)))* as1_v4i_p;
typedef v4i_ls __attribute__((address_space(3)))* as3_v4i_p;

// Async 16B global -> LDS copy (per lane), tracked by ASYNCcnt.
__device__ __forceinline__ void async_b128(const void* g, void* l) {
  __builtin_amdgcn_global_load_async_to_lds_b128(
      (as1_v4i_p)(unsigned long long)g,
      (as3_v4i_p)(unsigned int)(unsigned long long)l,
      0, 0);
}
__device__ __forceinline__ void wait_asynccnt0() {
#if defined(__has_builtin) && __has_builtin(__builtin_amdgcn_s_wait_asynccnt)
  __builtin_amdgcn_s_wait_asynccnt(0);
#else
  asm volatile("s_wait_asynccnt 0x0" ::: "memory");
#endif
}
#endif

// ---------------------------------------------------------------------------
// Kernel 0 (prep): fp32 W[k][n] -> bf16 Wt[n][k], for 6 weight matrices.
// wt_all[m] is 128x128 bf16, row = output channel n, contiguous in k.
// ---------------------------------------------------------------------------
__global__ __launch_bounds__(256)
void k0_prepw(const float* __restrict__ Wlp, const float* __restrict__ Wlg,
              const float* __restrict__ Wrp, const float* __restrict__ Wrg,
              const float* __restrict__ Wg,  const float* __restrict__ Wout,
              unsigned short* __restrict__ wt_all)
{
  __shared__ float tile[32][132];
  const int t = threadIdx.x;
  const float* W;
  switch (blockIdx.x) {
    case 0: W = Wlp; break;
    case 1: W = Wlg; break;
    case 2: W = Wrp; break;
    case 3: W = Wrg; break;
    case 4: W = Wg;  break;
    default: W = Wout; break;
  }
  unsigned int* dst = (unsigned int*)(wt_all + (size_t)blockIdx.x * 128 * 128);
  for (int kc = 0; kc < 4; ++kc) {
#pragma unroll
    for (int it = 0; it < 16; ++it) {         // load 32 rows of W, coalesced
      int idx = it * 256 + t;
      int kk = idx >> 7, n = idx & 127;
      tile[kk][n] = W[(kc * 32 + kk) * 128 + n];
    }
    __syncthreads();
#pragma unroll
    for (int it = 0; it < 8; ++it) {          // write transposed bf16 pairs, coalesced
      int idx = it * 256 + t;
      int n = idx >> 4, qq = idx & 15;
      dst[n * 64 + kc * 16 + qq] = pack2(tile[2 * qq][n], tile[2 * qq + 1][n]);
    }
    __syncthreads();
  }
}

// ---------------------------------------------------------------------------
// Kernel 1: LN(act) + 5 projections (lp, lg, rp, rg, g), fused gating.
// Outputs channel-major bf16: left_t[c][p], right_t[c][p], gate_t[c][p].
// Block = 16 positions, 256 threads (8 waves); wave w owns out channels 16w..16w+15.
// B-fragments loaded directly from pre-transposed bf16 weights (L2-resident).
// ---------------------------------------------------------------------------
__global__ __launch_bounds__(256)
void ka_proj(const float* __restrict__ act, const float* __restrict__ mask,
             const float* __restrict__ lnw, const float* __restrict__ lnb,
             const float* __restrict__ blp, const float* __restrict__ brp,
             const float* __restrict__ blg, const float* __restrict__ brg,
             const float* __restrict__ bg,
             const unsigned short* __restrict__ wt_all,
             unsigned short* __restrict__ left_t,
             unsigned short* __restrict__ right_t,
             unsigned short* __restrict__ gate_t)
{
  __shared__ float red0[16][16];
  __shared__ float red1[16][16];
  __shared__ float mu_s[16], rs_s[16], mask_s[16];
  __shared__ unsigned int x_u[16][66];     // 16 rows x 64 packed bf16 pairs (+pad)

  const int t = threadIdx.x;
  const int p_base = blockIdx.x * 16;
  const int r = t >> 4;
  const int q = t & 15;

  // ---- Stage 1: load 16x128 act rows, layer-norm, pack bf16 into LDS ----
  float v[8];
  const float* arow = act + (size_t)(p_base + r) * CZ + q * 8;
  float s = 0.f, s2 = 0.f;
#pragma unroll
  for (int e = 0; e < 8; ++e) { v[e] = arow[e]; s += v[e]; s2 += v[e] * v[e]; }
  red0[r][q] = s; red1[r][q] = s2;
  if (t < 16) mask_s[t] = mask[p_base + t];
  __syncthreads();
  if (t < 16) {
    float a = 0.f, b = 0.f;
#pragma unroll
    for (int e = 0; e < 16; ++e) { a += red0[t][e]; b += red1[t][e]; }
    float mu = a * (1.0f / CZ);
    float var = b * (1.0f / CZ) - mu * mu;
    mu_s[t] = mu; rs_s[t] = rsqrtf(var + 1e-5f);
  }
  __syncthreads();
  {
    float mu = mu_s[r], rs = rs_s[r];
#pragma unroll
    for (int e = 0; e < 8; e += 2) {
      int c = q * 8 + e;
      float y0 = (v[e]     - mu) * rs * lnw[c]     + lnb[c];
      float y1 = (v[e + 1] - mu) * rs * lnw[c + 1] + lnb[c + 1];
      x_u[r][c >> 1] = pack2(y0, y1);
    }
  }
  __syncthreads();

  const int wv   = t >> 5;
  const int lane = t & 31;
  const int ln16 = lane & 15;
  const int hi   = lane >> 4;
  const int col  = wv * 16 + ln16;   // this lane's output channel

  // A fragments (x) are shared by all 5 projections: build once.
  FragBF afrag[4];
#pragma unroll
  for (int ks = 0; ks < 4; ++ks) {
#pragma unroll
    for (int vv = 0; vv < 8; ++vv) {
      int uidx = vv + (hi ? 8 : 0);  // A K-pair layout
      afrag[ks].u[vv] = x_u[ln16][ks * 16 + uidx];
    }
  }

  v8f acc[5];
#pragma unroll
  for (int pj = 0; pj < 5; ++pj) {
    const unsigned int* wt_u = (const unsigned int*)(wt_all + (size_t)pj * 128 * 128);
    v8f c = {};
#pragma unroll
    for (int ks = 0; ks < 4; ++ks) {
      FragBF bfrag;   // 8 uints = 32 contiguous bytes of Wt[col][.]
      const uint4* wp = (const uint4*)(wt_u + col * 64 + ks * 16 + hi * 8);
      bfrag.q[0] = wp[0];
      bfrag.q[1] = wp[1];
      c = __builtin_amdgcn_wmma_f32_16x16x32_bf16(false, afrag[ks].v,
                                                  false, bfrag.v,
                                                  (short)0, c, false, false);
    }
    acc[pj] = c;
  }

  // ---- Epilogue: bias, mask*proj*sigmoid(gate-proj), pack & store ----
  float b0 = blp[col], b1 = blg[col], b2 = brp[col], b3 = brg[col], b4 = bg[col];
  unsigned int lres[4], rres[4], gres[4];
#pragma unroll
  for (int vv = 0; vv < 8; vv += 2) {
    int r0 = vv + hi * 8, r1 = r0 + 1;
    float m0 = mask_s[r0], m1 = mask_s[r1];
    float l0 = m0 * (acc[0][vv]     + b0) * sigmoidf_(acc[1][vv]     + b1);
    float l1 = m1 * (acc[0][vv + 1] + b0) * sigmoidf_(acc[1][vv + 1] + b1);
    float q0 = m0 * (acc[2][vv]     + b2) * sigmoidf_(acc[3][vv]     + b3);
    float q1 = m1 * (acc[2][vv + 1] + b2) * sigmoidf_(acc[3][vv + 1] + b3);
    float g0 = sigmoidf_(acc[4][vv]     + b4);
    float g1 = sigmoidf_(acc[4][vv + 1] + b4);
    lres[vv >> 1] = pack2(l0, l1);
    rres[vv >> 1] = pack2(q0, q1);
    gres[vv >> 1] = pack2(g0, g1);
  }
  size_t off = (size_t)col * NN + p_base + hi * 8;   // 16B-aligned
  uint4 u;
  u.x = lres[0]; u.y = lres[1]; u.z = lres[2]; u.w = lres[3];
  *(uint4*)(left_t + off) = u;
  u.x = rres[0]; u.y = rres[1]; u.z = rres[2]; u.w = rres[3];
  *(uint4*)(right_t + off) = u;
  u.x = gres[0]; u.y = gres[1]; u.z = gres[2]; u.w = gres[3];
  *(uint4*)(gate_t + off) = u;
}

// ---------------------------------------------------------------------------
// Kernel 2: batched NT GEMM per channel: ein_t[c] = L_c (768x768) @ R_c^T.
// Grid (6,6,128); block computes a 128x128 tile; 8 waves of 64x32 each.
// Double-buffered LDS tiles; async global->LDS copies (ASYNCcnt) when available.
// ---------------------------------------------------------------------------
__global__ __launch_bounds__(256)
void kb_einsum(const unsigned short* __restrict__ left_t,
               const unsigned short* __restrict__ right_t,
               float* __restrict__ ein_t)
{
  __shared__ unsigned int As[2][128 * 20];  // 128 rows x 16 K-pair uints (+4 pad)
  __shared__ unsigned int Bs[2][128 * 20];

  const int t = threadIdx.x;
  const int ibase = blockIdx.x * 128;
  const int jbase = blockIdx.y * 128;
  const int c = blockIdx.z;
  const size_t base = (size_t)c * NN;

  const int lane = t & 31;
  const int wv = t >> 5;
  const int wm = (wv >> 2) * 64;   // 0 / 64
  const int wn = (wv & 3) * 32;    // 0 / 32 / 64 / 96
  const int ln16 = lane & 15;
  const int hi = lane >> 4;

  v8f acc[4][2] = {};

  const int lrow = t >> 1;
  const int lhalf = t & 1;
  const unsigned short* aptr = left_t  + base + (size_t)(ibase + lrow) * NRES + lhalf * 16;
  const unsigned short* bptr = right_t + base + (size_t)(jbase + lrow) * NRES + lhalf * 16;
  const int lidx = lrow * 20 + lhalf * 8;

  // Prologue: tile 0 into buffer 0.
#if HAVE_ASYNC_LDS
  async_b128(aptr,     &As[0][lidx]);
  async_b128(aptr + 8, &As[0][lidx + 4]);
  async_b128(bptr,     &Bs[0][lidx]);
  async_b128(bptr + 8, &Bs[0][lidx + 4]);
  wait_asynccnt0();
#else
  {
    const uint4* ag = (const uint4*)aptr;
    const uint4* bg = (const uint4*)bptr;
    ((uint4*)&As[0][lidx])[0] = ag[0]; ((uint4*)&As[0][lidx + 4])[0] = ag[1];
    ((uint4*)&Bs[0][lidx])[0] = bg[0]; ((uint4*)&Bs[0][lidx + 4])[0] = bg[1];
  }
#endif

  for (int kb = 0; kb < 24; ++kb) {
    const int cur = kb & 1;
    __syncthreads();                         // buf[cur] ready; buf[cur^1] free

#if HAVE_ASYNC_LDS
    if (kb + 1 < 24) {                       // async-copy next tile into free buffer
      const unsigned short* ga = aptr + (kb + 1) * 32;
      const unsigned short* gb = bptr + (kb + 1) * 32;
      async_b128(ga,     &As[cur ^ 1][lidx]);
      async_b128(ga + 8, &As[cur ^ 1][lidx + 4]);
      async_b128(gb,     &Bs[cur ^ 1][lidx]);
      async_b128(gb + 8, &Bs[cur ^ 1][lidx + 4]);
      if (kb + 2 < 24) {
        __builtin_prefetch(aptr + (kb + 2) * 32, 0, 0);
        __builtin_prefetch(bptr + (kb + 2) * 32, 0, 0);
      }
    }
#else
    uint4 ra0, ra1, rb0, rb1;
    if (kb + 1 < 24) {                       // issue next tile's loads early
      const uint4* ag = (const uint4*)(aptr + (kb + 1) * 32);
      const uint4* bg = (const uint4*)(bptr + (kb + 1) * 32);
      ra0 = ag[0]; ra1 = ag[1];
      rb0 = bg[0]; rb1 = bg[1];
      if (kb + 2 < 24) {
        __builtin_prefetch(aptr + (kb + 2) * 32, 0, 0);
        __builtin_prefetch(bptr + (kb + 2) * 32, 0, 0);
      }
    }
#endif

    FragBF af[4], bf[2];
#pragma unroll
    for (int mt = 0; mt < 4; ++mt) {
      int row = wm + mt * 16 + ln16;
#pragma unroll
      for (int vv = 0; vv < 8; ++vv) {
        int uidx = vv + (hi ? 8 : 0);
        af[mt].u[vv] = As[cur][row * 20 + uidx];
      }
    }
#pragma unroll
    for (int nt = 0; nt < 2; ++nt) {
      int row = wn + nt * 16 + ln16;
#pragma unroll
      for (int vv = 0; vv < 8; ++vv)
        bf[nt].u[vv] = Bs[cur][row * 20 + (hi ? 8 : 0) + vv];
    }
#pragma unroll
    for (int mt = 0; mt < 4; ++mt)
#pragma unroll
      for (int nt = 0; nt < 2; ++nt)
        acc[mt][nt] = __builtin_amdgcn_wmma_f32_16x16x32_bf16(false, af[mt].v,
                                                              false, bf[nt].v,
                                                              (short)0, acc[mt][nt],
                                                              false, false);

#if HAVE_ASYNC_LDS
    if (kb + 1 < 24) wait_asynccnt0();       // copies done before next barrier
#else
    if (kb + 1 < 24) {                       // stash next tile into the free buffer
      ((uint4*)&As[cur ^ 1][lidx])[0] = ra0; ((uint4*)&As[cur ^ 1][lidx + 4])[0] = ra1;
      ((uint4*)&Bs[cur ^ 1][lidx])[0] = rb0; ((uint4*)&Bs[cur ^ 1][lidx + 4])[0] = rb1;
    }
#endif
  }

#pragma unroll
  for (int mt = 0; mt < 4; ++mt) {
#pragma unroll
    for (int nt = 0; nt < 2; ++nt) {
      int colj = jbase + wn + nt * 16 + ln16;
#pragma unroll
      for (int vv = 0; vv < 8; ++vv) {
        int row = ibase + wm + mt * 16 + vv + hi * 8;
        ein_t[base + (size_t)row * NRES + colj] = acc[mt][nt][vv];
      }
    }
  }
}

// ---------------------------------------------------------------------------
// Kernel 3: LN(ein) @ W_out + b_out, times stored sigmoid gate -> d_out.
// Block = 16 positions (same i, 16 consecutive j), 256 threads.
// ---------------------------------------------------------------------------
__global__ __launch_bounds__(256)
void kc_out(const float* __restrict__ ein_t,
            const unsigned short* __restrict__ gate_t,
            const float* __restrict__ lnw, const float* __restrict__ lnb,
            const unsigned short* __restrict__ wt_out,   // pre-transposed bf16 W_out
            const float* __restrict__ bout,
            float* __restrict__ out)
{
  __shared__ float xs[16][132];            // gathered rows; reused for result
  __shared__ unsigned short gs[16][130];
  __shared__ unsigned int x_u[16][66];
  __shared__ float red0[16][16];
  __shared__ float red1[16][16];
  __shared__ float mu_s[16], rs_s[16];

  const int t = threadIdx.x;
  const int p_base = blockIdx.x * 16;

  // Gather channel-major -> row-major (coalesced: 16 consecutive j per channel)
#pragma unroll
  for (int it = 0; it < 8; ++it) {
    int cch = it * 16 + (t >> 4);
    int jl = t & 15;
    size_t off = (size_t)cch * NN + p_base + jl;
    xs[jl][cch] = ein_t[off];
    gs[jl][cch] = gate_t[off];
  }
  __syncthreads();

  const int r = t >> 4, q = t & 15;
  float v[8]; float s = 0.f, s2 = 0.f;
#pragma unroll
  for (int e = 0; e < 8; ++e) { v[e] = xs[r][q * 8 + e]; s += v[e]; s2 += v[e] * v[e]; }
  red0[r][q] = s; red1[r][q] = s2;
  __syncthreads();
  if (t < 16) {
    float a = 0.f, b = 0.f;
#pragma unroll
    for (int e = 0; e < 16; ++e) { a += red0[t][e]; b += red1[t][e]; }
    float mu = a * (1.0f / CZ);
    float var = b * (1.0f / CZ) - mu * mu;
    mu_s[t] = mu; rs_s[t] = rsqrtf(var + 1e-5f);
  }
  __syncthreads();
  {
    float mu = mu_s[r], rs = rs_s[r];
#pragma unroll
    for (int e = 0; e < 8; e += 2) {
      int cc = q * 8 + e;
      float y0 = (v[e]     - mu) * rs * lnw[cc]     + lnb[cc];
      float y1 = (v[e + 1] - mu) * rs * lnw[cc + 1] + lnb[cc + 1];
      x_u[r][cc >> 1] = pack2(y0, y1);
    }
  }
  __syncthreads();

  const int wv = t >> 5;
  const int lane = t & 31;
  const int ln16 = lane & 15;
  const int hi = lane >> 4;
  const int col = wv * 16 + ln16;

  const unsigned int* wt_u = (const unsigned int*)wt_out;
  v8f cacc = {};
#pragma unroll
  for (int ks = 0; ks < 4; ++ks) {
    FragBF afrag, bfrag;
#pragma unroll
    for (int vv = 0; vv < 8; ++vv) {
      int uidx = vv + (hi ? 8 : 0);
      afrag.u[vv] = x_u[ln16][ks * 16 + uidx];
    }
    const uint4* wp = (const uint4*)(wt_u + col * 64 + ks * 16 + hi * 8);
    bfrag.q[0] = wp[0];
    bfrag.q[1] = wp[1];
    cacc = __builtin_amdgcn_wmma_f32_16x16x32_bf16(false, afrag.v, false, bfrag.v,
                                                   (short)0, cacc, false, false);
  }

  float bias = bout[col];
#pragma unroll
  for (int vv = 0; vv < 8; ++vv) {
    int row = vv + hi * 8;
    float g = bf2f(gs[row][col]);
    xs[row][col] = (cacc[vv] + bias) * g;   // stage result in LDS (unique slot/lane)
  }
  __syncthreads();
#pragma unroll
  for (int it = 0; it < 8; ++it) {
    int idx = it * 256 + t;
    int rr = idx >> 7, cc = idx & 127;
    out[(size_t)(p_base + rr) * CZ + cc] = xs[rr][cc];
  }
}

// ---------------------------------------------------------------------------
extern "C" void kernel_launch(void* const* d_in, const int* in_sizes, int n_in,
                              void* d_out, int out_size, void* d_ws, size_t ws_size,
                              hipStream_t stream) {
  (void)in_sizes; (void)n_in; (void)out_size; (void)ws_size;
  const float* act     = (const float*)d_in[0];
  const float* mask    = (const float*)d_in[1];
  const float* ln_in_w = (const float*)d_in[2];
  const float* ln_in_b = (const float*)d_in[3];
  const float* W_lp = (const float*)d_in[4];
  const float* b_lp = (const float*)d_in[5];
  const float* W_rp = (const float*)d_in[6];
  const float* b_rp = (const float*)d_in[7];
  const float* W_lg = (const float*)d_in[8];
  const float* b_lg = (const float*)d_in[9];
  const float* W_rg = (const float*)d_in[10];
  const float* b_rg = (const float*)d_in[11];
  const float* ln_c_w = (const float*)d_in[12];
  const float* ln_c_b = (const float*)d_in[13];
  const float* W_out  = (const float*)d_in[14];
  const float* b_out  = (const float*)d_in[15];
  const float* W_g    = (const float*)d_in[16];
  const float* b_g    = (const float*)d_in[17];

  char* ws = (char*)d_ws;
  const size_t szbf = (size_t)NN * CZ * 2;             // 151 MB per bf16 plane set
  unsigned short* left_t  = (unsigned short*)(ws);
  unsigned short* right_t = (unsigned short*)(ws + szbf);
  unsigned short* gate_t  = (unsigned short*)(ws + 2 * szbf);
  float*          ein_t   = (float*)(ws + 3 * szbf);   // 302 MB fp32
  unsigned short* wt_all  = (unsigned short*)(ws + 3 * szbf + (size_t)NN * CZ * 4);

  k0_prepw<<<6, 256, 0, stream>>>(W_lp, W_lg, W_rp, W_rg, W_g, W_out, wt_all);
  ka_proj<<<NN / 16, 256, 0, stream>>>(act, mask, ln_in_w, ln_in_b,
                                       b_lp, b_rp, b_lg, b_rg, b_g,
                                       wt_all, left_t, right_t, gate_t);
  kb_einsum<<<dim3(NRES / 128, NRES / 128, CZ), 256, 0, stream>>>(left_t, right_t, ein_t);
  kc_out<<<NN / 16, 256, 0, stream>>>(ein_t, gate_t, ln_c_w, ln_c_b,
                                      wt_all + (size_t)5 * 128 * 128, b_out,
                                      (float*)d_out);
}
